// GCN_88802743812362
// MI455X (gfx1250) — matrix operand order repaired and checked by reading, not compiled, not run on hardware
//
#include <hip/hip_runtime.h>

typedef __attribute__((ext_vector_type(16))) _Float16 v16h;
typedef __attribute__((ext_vector_type(8)))  _Float16 h8;
typedef __attribute__((ext_vector_type(8)))  float    v8f;

#define N_NODES  100000
#define N_EDGES  3200000
#define D_FEAT   128
#define HIDDEN   64
#define N_GRAPHS 64

static __device__ __forceinline__ void atomicAddF(float* p, float v) {
  __hip_atomic_fetch_add(p, v, __ATOMIC_RELAXED, __HIP_MEMORY_SCOPE_AGENT);
}
static __device__ __forceinline__ void atomicAddU(unsigned* p, unsigned v) {
  __hip_atomic_fetch_add(p, v, __ATOMIC_RELAXED, __HIP_MEMORY_SCOPE_AGENT);
}

// ---------------- utility kernels ----------------

__global__ void k_zero(unsigned* p, long long n) {
  long long i = (long long)blockIdx.x * blockDim.x + threadIdx.x;
  if (i < n) p[i] = 0u;
}

__global__ void k_degcnt(const int* __restrict__ dst, unsigned* deg) {
  int e = blockIdx.x * blockDim.x + threadIdx.x;
  if (e < N_EDGES) atomicAddU(&deg[dst[e]], 1u);
}

// in-place: buffer holds uint counts, replaced by rsqrt(deg+1) as float
__global__ void k_dis(float* dis) {
  int i = blockIdx.x * blockDim.x + threadIdx.x;
  if (i < N_NODES) {
    unsigned c = ((const unsigned*)dis)[i];
    dis[i] = rsqrtf((float)c + 1.0f);
  }
}

__global__ void k_cvt(const float* __restrict__ in, _Float16* __restrict__ out, long long n) {
  long long i = (long long)blockIdx.x * blockDim.x + threadIdx.x;
  if (i < n) out[i] = (_Float16)in[i];
}

// Pack weight matrix W [K x 64] (f32, row-major) into this file's B-fragment order (contiguous k, same as A):
// frag (nt, ks): lane&15 = N within tile, lane>>4 selects K+16 half,
// VGPR v holds K pair (2v, 2v+1). 16 halves per lane, contiguous.
__global__ void k_prepB(const float* __restrict__ W, _Float16* __restrict__ Bf, int K) {
  int KS = K >> 5;
  int t = blockIdx.x * blockDim.x + threadIdx.x;
  if (t >= 4 * KS * 32) return;
  int lane = t & 31;
  int ks = (t >> 5) % KS;
  int nt = (t >> 5) / KS;
  int n  = nt * 16 + (lane & 15);
  int kb = ks * 32 + (lane >> 4) * 16;
  _Float16* o = Bf + ((size_t)(nt * KS + ks) * 32 + lane) * 16;
#pragma unroll
  for (int v = 0; v < 8; ++v) {
    o[2 * v]     = (_Float16)W[(size_t)(kb + 2 * v)     * HIDDEN + n];
    o[2 * v + 1] = (_Float16)W[(size_t)(kb + 2 * v + 1) * HIDDEN + n];
  }
}

// ---------------- WMMA GEMM: T[M x 64] = A[M x K](f16) @ W  (+bias, relu) ----------------
// One wave per 16-row M tile; 4 N tiles of 16; K in steps of 32.
__global__ __launch_bounds__(256) void k_gemm_wmma(
    const _Float16* A, const _Float16* __restrict__ Bf,
    float* T, _Float16* Th, const float* __restrict__ bias, int K, int relu)
{
  const int KS = K >> 5;
  int gwave = (int)(((long long)blockIdx.x * blockDim.x + threadIdx.x) >> 5);
  if (gwave >= N_NODES / 16) return;     // wave-uniform exit: EXEC stays all-ones
  int lane = threadIdx.x & 31;
  int m  = lane & 15;
  int kg = lane >> 4;
  const _Float16* Ap = A + (size_t)(gwave * 16 + m) * K;

  v8f c[4] = {};
  for (int ks = 0; ks < KS; ++ks) {
    int k0 = ks * 32 + kg * 16;
    h8 alo = *(const h8*)(Ap + k0);        // VGPRs 0-3: K = k0..k0+7
    h8 ahi = *(const h8*)(Ap + k0 + 8);    // VGPRs 4-7: K = k0+8..k0+15
    v16h a;
#pragma unroll
    for (int i = 0; i < 8; ++i) { a[i] = alo[i]; a[i + 8] = ahi[i]; }
#pragma unroll
    for (int nt = 0; nt < 4; ++nt) {
      const h8* bp = (const h8*)(Bf + ((size_t)(nt * KS + ks) * 32 + lane) * 16);
      h8 blo = bp[0], bhi = bp[1];
      v16h b;
#pragma unroll
      for (int i = 0; i < 8; ++i) { b[i] = blo[i]; b[i + 8] = bhi[i]; }
      c[nt] = __builtin_amdgcn_wmma_f32_16x16x32_f16(
          false, a, false, b, (short)0, c[nt], false, false);
    }
  }

  int n0 = lane & 15;
#pragma unroll
  for (int r = 0; r < 8; ++r) {
    int row = gwave * 16 + kg * 8 + r;     // C layout: lanes>=16 hold M = r + 8
#pragma unroll
    for (int nt = 0; nt < 4; ++nt) {
      float v = c[nt][r];
      int col = nt * 16 + n0;
      if (bias) v += bias[col];
      if (relu) v = fmaxf(v, 0.0f);
      T[(size_t)row * HIDDEN + col] = v;
      if (Th) Th[(size_t)row * HIDDEN + col] = (_Float16)v;
    }
  }
}

// ---------------- edge scatter: AGG[dst] += dis[src]*dis[dst]*T[src] ----------------
__global__ void k_scatter(const int* __restrict__ src, const int* __restrict__ dst,
                          const float* __restrict__ dis, const float* __restrict__ T,
                          float* AGG) {
  long long i = (long long)blockIdx.x * blockDim.x + threadIdx.x;
  if (i >= (long long)N_EDGES * HIDDEN) return;
  int e = (int)(i >> 6);
  int c = (int)(i & 63);
  int s = src[e], d = dst[e];
  float coef = dis[s] * dis[d];
  atomicAddF(&AGG[(size_t)d * HIDDEN + c], coef * T[(size_t)s * HIDDEN + c]);
}

// H = relu(AGG + dis^2 * T + bias), in place in AGG; optional f16 copy
__global__ void k_finish(float* AGG, const float* __restrict__ T,
                         const float* __restrict__ dis, const float* __restrict__ bias,
                         _Float16* Hh) {
  long long i = (long long)blockIdx.x * blockDim.x + threadIdx.x;
  if (i >= (long long)N_NODES * HIDDEN) return;
  int n = (int)(i >> 6);
  int c = (int)(i & 63);
  float ds = dis[n];
  float v = AGG[i] + ds * ds * T[i] + bias[c];
  v = fmaxf(v, 0.0f);
  AGG[i] = v;
  if (Hh) Hh[i] = (_Float16)v;
}

// ---------------- pooling + readout ----------------
__global__ void k_pool(const float* __restrict__ H, const int* __restrict__ batch,
                       float* pooled, unsigned* counts) {
  long long i = (long long)blockIdx.x * blockDim.x + threadIdx.x;
  if (i >= (long long)N_NODES * HIDDEN) return;
  int n = (int)(i >> 6);
  int c = (int)(i & 63);
  int g = batch[n];
  atomicAddF(&pooled[(size_t)g * HIDDEN + c], H[i]);
  if (c == 0) atomicAddU(&counts[g], 1u);
}

__global__ void k_final(const float* __restrict__ pooled, const unsigned* __restrict__ counts,
                        const float* __restrict__ linW, const float* __restrict__ linb,
                        float* out) {
  int g = threadIdx.x;
  if (g >= N_GRAPHS) return;
  float cnt = fmaxf((float)counts[g], 1.0f);
  float acc = 0.0f;
  for (int c = 0; c < HIDDEN; ++c) acc += pooled[(size_t)g * HIDDEN + c] * linW[c];
  out[g] = acc / cnt + linb[0];
}

// ---------------- host launch ----------------
extern "C" void kernel_launch(void* const* d_in, const int* in_sizes, int n_in,
                              void* d_out, int out_size, void* d_ws, size_t ws_size,
                              hipStream_t stream) {
  (void)in_sizes; (void)n_in; (void)out_size; (void)ws_size;
  const float* x    = (const float*)d_in[0];
  const int*   ei   = (const int*)d_in[1];
  const int*   batch= (const int*)d_in[2];
  const float* W1   = (const float*)d_in[3];
  const float* b1   = (const float*)d_in[4];
  const float* mlpW = (const float*)d_in[5];
  const float* mlpb = (const float*)d_in[6];
  const float* W2   = (const float*)d_in[7];
  const float* b2   = (const float*)d_in[8];
  const float* W3   = (const float*)d_in[9];
  const float* b3   = (const float*)d_in[10];
  const float* linW = (const float*)d_in[11];
  const float* linb = (const float*)d_in[12];
  float* out = (float*)d_out;
  const int* src = ei;
  const int* dst = ei + N_EDGES;

  char* w = (char*)d_ws;
  auto carve = [&](size_t bytes) -> char* {
    char* p = w; w += (bytes + 255) & ~(size_t)255; return p;
  };
  float*    dis    = (float*)carve((size_t)N_NODES * 4);
  float*    T      = (float*)carve((size_t)N_NODES * HIDDEN * 4);
  float*    AGG    = (float*)carve((size_t)N_NODES * HIDDEN * 4);
  _Float16* Ah     = (_Float16*)carve((size_t)N_NODES * D_FEAT * 2); // also f16 H buffer
  _Float16* Bf     = (_Float16*)carve((size_t)4 * 4 * 32 * 16 * 2);
  float*    pooled = (float*)carve((size_t)N_GRAPHS * HIDDEN * 4);
  unsigned* counts = (unsigned*)carve((size_t)N_GRAPHS * 4);

  const dim3 B(256);
  auto blk = [](long long n) { return dim3((unsigned)((n + 255) / 256)); };
  const dim3 gemmGrid((N_NODES / 16 + 7) / 8);   // 8 waves per 256-thread block
  const long long NH = (long long)N_NODES * HIDDEN;
  const long long EH = (long long)N_EDGES * HIDDEN;

  // degrees: deg = segment_sum(1 over dst) + 1 ; dis = rsqrt(deg)
  k_zero<<<blk(N_NODES), B, 0, stream>>>((unsigned*)dis, N_NODES);
  k_degcnt<<<blk(N_EDGES), B, 0, stream>>>(dst, (unsigned*)dis);
  k_dis<<<blk(N_NODES), B, 0, stream>>>(dis);

  // ---- conv1: T = x@W1 ; AGG = scatter ; H = relu(AGG + dis^2*T + b1)
  k_cvt<<<blk((long long)N_NODES * D_FEAT), B, 0, stream>>>(x, Ah, (long long)N_NODES * D_FEAT);
  k_prepB<<<blk(4 * (D_FEAT / 32) * 32), B, 0, stream>>>(W1, Bf, D_FEAT);
  k_gemm_wmma<<<gemmGrid, B, 0, stream>>>(Ah, Bf, T, nullptr, nullptr, D_FEAT, 0);
  k_zero<<<blk(NH), B, 0, stream>>>((unsigned*)AGG, NH);
  k_scatter<<<blk(EH), B, 0, stream>>>(src, dst, dis, T, AGG);
  k_finish<<<blk(NH), B, 0, stream>>>(AGG, T, dis, b1, Ah);

  // ---- mlp: H = relu(H@mlpW + mlpb)  (f16 copy written back into Ah)
  k_prepB<<<blk(4 * (HIDDEN / 32) * 32), B, 0, stream>>>(mlpW, Bf, HIDDEN);
  k_gemm_wmma<<<gemmGrid, B, 0, stream>>>(Ah, Bf, T, Ah, mlpb, HIDDEN, 1);

  // ---- conv2
  k_prepB<<<blk(4 * (HIDDEN / 32) * 32), B, 0, stream>>>(W2, Bf, HIDDEN);
  k_gemm_wmma<<<gemmGrid, B, 0, stream>>>(Ah, Bf, T, nullptr, nullptr, HIDDEN, 0);
  k_zero<<<blk(NH), B, 0, stream>>>((unsigned*)AGG, NH);
  k_scatter<<<blk(EH), B, 0, stream>>>(src, dst, dis, T, AGG);
  k_finish<<<blk(NH), B, 0, stream>>>(AGG, T, dis, b2, Ah);

  // ---- conv3
  k_prepB<<<blk(4 * (HIDDEN / 32) * 32), B, 0, stream>>>(W3, Bf, HIDDEN);
  k_gemm_wmma<<<gemmGrid, B, 0, stream>>>(Ah, Bf, T, nullptr, nullptr, HIDDEN, 0);
  k_zero<<<blk(NH), B, 0, stream>>>((unsigned*)AGG, NH);
  k_scatter<<<blk(EH), B, 0, stream>>>(src, dst, dis, T, AGG);
  k_finish<<<blk(NH), B, 0, stream>>>(AGG, T, dis, b3, nullptr);

  // ---- global mean pool + linear readout
  k_zero<<<blk(N_GRAPHS * HIDDEN), B, 0, stream>>>((unsigned*)pooled, N_GRAPHS * HIDDEN);
  k_zero<<<blk(N_GRAPHS), B, 0, stream>>>(counts, N_GRAPHS);
  k_pool<<<blk(NH), B, 0, stream>>>(AGG, batch, pooled, counts);
  k_final<<<dim3(1), dim3(64), 0, stream>>>(pooled, counts, linW, linb, out);
}
